// SpatialGroupConv3dTestTorch_71313636983476
// MI455X (gfx1250) — compile-verified, matrix operand order checked by reading, not run on hardware
//
#include <hip/hip_runtime.h>

// ---- WMMA vector types (CDNA5 / gfx1250, wave32) ----
typedef __attribute__((ext_vector_type(16))) __bf16 v16bf;
typedef __attribute__((ext_vector_type(8)))  float  v8f;
typedef __attribute__((ext_vector_type(4)))  float  f4;
typedef __attribute__((ext_vector_type(4)))  int    i4;

// address-space qualified int4 for the async global->LDS builtin
typedef __attribute__((address_space(1))) i4 gi4;   // global
typedef __attribute__((address_space(3))) i4 li4;   // LDS

// ---- problem constants (fixed by the reference) ----
#define DD    41
#define HH    400
#define WW    400
#define BATCH 2
#define GRID_CELLS (BATCH * DD * HH * WW)   // 13,120,000 ints = 52.48 MB
#define CIN   64
#define COUT  64
#define NPOS  27                            // 3^3 kernel positions
#define WPACK_TILES (NPOS * 2 * 4)          // p * kstep * ntile = 216
#define WPACK_ELEMS (WPACK_TILES * 32)      // 6912 v16bf fragments
#define WPACK_BYTES (WPACK_ELEMS * 32)      // 221,184 B  (< 320 KB WGP LDS)

#define CONV_THREADS 512                    // 16 waves / workgroup

// ======================================================================
// Kernel 1: dense voxel grid <- -1
// ======================================================================
__global__ void sgc_init_grid(int* __restrict__ grid) {
    const int n4 = GRID_CELLS / 4;
    int i = blockIdx.x * blockDim.x + threadIdx.x;
    if (i < n4) {
        i4 fill = { -1, -1, -1, -1 };
        ((i4*)grid)[i] = fill;
    }
}

// ======================================================================
// Kernel 2: scatter voxel index into grid
// ======================================================================
__global__ void sgc_scatter(const int* __restrict__ coors,
                            int* __restrict__ grid, int N) {
    int n = blockIdx.x * blockDim.x + threadIdx.x;
    if (n >= N) return;
    i4 c = ((const i4*)coors)[n];   // (b, z, y, x)
    grid[((c.x * DD + c.y) * HH + c.z) * WW + c.w] = n;
}

// ======================================================================
// Kernel 3: pre-pack weights into bf16 WMMA B-matrix lane layout.
// Tile id = p*8 + t*4 + j.  B fragment layout (16-bit, 32x16, wave32):
//   lane L holds column n = L&15; kb = (L>>4)*8
//   elements 0..7  = B[kb+0 .. kb+7][n]
//   elements 8..15 = B[16+kb+0 .. 16+kb+7][n]
// with B[k][n] = weight[g][32t + k][16j + n],  g = k2g[p].
// ======================================================================
__global__ void sgc_pack_weights(const float* __restrict__ weight,
                                 const int* __restrict__ gmap,
                                 v16bf* __restrict__ wpack) {
    const int tile = blockIdx.x;           // 0..215
    const int p = tile >> 3;
    const int t = (tile >> 2) & 1;
    const int j = tile & 3;

    // invert group_map: find g with p in group_map[g][:]
    int g = 0;
    for (int gg = 0; gg < NPOS; ++gg)
        for (int s = 0; s < NPOS; ++s)
            if (gmap[gg * NPOS + s] == p) g = gg;

    const int L  = threadIdx.x;            // 0..31
    const int n  = L & 15;
    const int kb = (L >> 4) * 8;

    v16bf v;
#pragma unroll
    for (int e = 0; e < 16; ++e) {
        const int k   = (e < 8) ? (kb + e) : (16 + kb + (e - 8));
        const int cin = 32 * t + k;
        v[e] = (__bf16)weight[(g * CIN + cin) * COUT + (16 * j + n)];
    }
    wpack[tile * 32 + L] = v;
}

// ======================================================================
// Kernel 4: main sparse group conv with v_wmma_f32_16x16x32_bf16.
// One wave handles 32 voxels (2 M-tiles of 16), all 64 output channels
// (4 N-tiles).  Full packed-weight set staged in LDS once per workgroup
// via CDNA5 async global->LDS, then B fragments come from ds_load
// instead of L2.
// ======================================================================
__global__ __launch_bounds__(CONV_THREADS)
void sgc_conv_wmma(const float* __restrict__ feat,
                   const int*   __restrict__ coors,
                   const v16bf* __restrict__ wpack,
                   const int*   __restrict__ grid,
                   float*       __restrict__ out,
                   int N, int nWaves) {
    __shared__ v16bf shW[WPACK_ELEMS];     // 221,184 B of LDS

    // ---- stage packed weights into LDS (all threads participate) ----
    {
        const int tid = threadIdx.x;
#if __has_builtin(__builtin_amdgcn_global_load_async_to_lds_b128) && \
    __has_builtin(__builtin_amdgcn_s_wait_asynccnt)
        const char* gsrc = (const char*)wpack;
        char*       ldst = (char*)&shW[0];
        for (int i = tid; i < WPACK_BYTES / 16; i += CONV_THREADS) {
            __builtin_amdgcn_global_load_async_to_lds_b128(
                (gi4*)(gsrc + (size_t)i * 16),
                (li4*)(ldst + (size_t)i * 16), 0, 0);
        }
        __builtin_amdgcn_s_wait_asynccnt(0);
#else
        for (int i = tid; i < WPACK_ELEMS; i += CONV_THREADS)
            shW[i] = wpack[i];
#endif
        __syncthreads();
    }

    const int wid = blockIdx.x * (CONV_THREADS / 32) + (threadIdx.x >> 5);
    if (wid >= nWaves) return;             // uniform per wave; after barrier

    const int L    = threadIdx.x & 31;
    const int m    = L & 15;               // A row / C column within tile
    const int half = L >> 4;               // 0 or 1
    const int kb   = half * 8;
    const int base = wid * 32;

    // voxel coordinates for both M-tiles (lanes 0-15 / 16-31 duplicate)
    i4 c[2];
#pragma unroll
    for (int a = 0; a < 2; ++a) {
        int v = base + 16 * a + m;
        v = (v < N) ? v : (N - 1);
        c[a] = ((const i4*)coors)[v];
    }

    const v8f vzero = {0.f, 0.f, 0.f, 0.f, 0.f, 0.f, 0.f, 0.f};
    v8f acc[2][4];
#pragma unroll
    for (int a = 0; a < 2; ++a)
#pragma unroll
        for (int j = 0; j < 4; ++j) acc[a][j] = vzero;

    for (int p = 0; p < NPOS; ++p) {
        const int dz = p / 9 - 1;
        const int dy = (p / 3) % 3 - 1;
        const int dx = p % 3 - 1;

        int  rowidx[2];
        bool val[2];
#pragma unroll
        for (int a = 0; a < 2; ++a) {
            const int nz = c[a].y + dz, ny = c[a].z + dy, nx = c[a].w + dx;
            const bool ok = ((unsigned)nz < (unsigned)DD) &
                            ((unsigned)ny < (unsigned)HH) &
                            ((unsigned)nx < (unsigned)WW);
            const int id = ok ? grid[((c[a].x * DD + nz) * HH + ny) * WW + nx] : -1;
            val[a]    = id >= 0;
            rowidx[a] = val[a] ? id : 0;
        }

#pragma unroll
        for (int t = 0; t < 2; ++t) {
            // Gather A fragments: per lane, two contiguous 8-channel runs
            // (aligned float4 pairs) -> bf16.
            v16bf A[2];
#pragma unroll
            for (int a = 0; a < 2; ++a) {
                const float* row = feat + (size_t)rowidx[a] * CIN;
                const int ch0 = 32 * t + kb;       // {0,8,32,40}: 16B aligned
                const int ch1 = ch0 + 16;
                f4 f0 = {0.f,0.f,0.f,0.f}, f1 = f0, f2 = f0, f3 = f0;
                if (val[a]) {
                    f0 = *(const f4*)(row + ch0);
                    f1 = *(const f4*)(row + ch0 + 4);
                    f2 = *(const f4*)(row + ch1);
                    f3 = *(const f4*)(row + ch1 + 4);
                }
                v16bf av;
                av[0]=(__bf16)f0.x; av[1]=(__bf16)f0.y; av[2]=(__bf16)f0.z; av[3]=(__bf16)f0.w;
                av[4]=(__bf16)f1.x; av[5]=(__bf16)f1.y; av[6]=(__bf16)f1.z; av[7]=(__bf16)f1.w;
                av[8]=(__bf16)f2.x; av[9]=(__bf16)f2.y; av[10]=(__bf16)f2.z; av[11]=(__bf16)f2.w;
                av[12]=(__bf16)f3.x; av[13]=(__bf16)f3.y; av[14]=(__bf16)f3.z; av[15]=(__bf16)f3.w;
                A[a] = av;
            }

            const v16bf* wp = &shW[((p * 2 + t) * 4) * 32 + L];
#pragma unroll
            for (int j = 0; j < 4; ++j) {
                const v16bf B = wp[j * 32];        // ds_load, reused for both M-tiles
                acc[0][j] = __builtin_amdgcn_wmma_f32_16x16x32_bf16(
                    false, A[0], false, B, (short)0, acc[0][j], false, false);
                acc[1][j] = __builtin_amdgcn_wmma_f32_16x16x32_bf16(
                    false, A[1], false, B, (short)0, acc[1][j], false, false);
            }
        }
    }

    // C/D layout: lane L, vgpr v -> row (v + 8*half), col (L&15) of tile.
#pragma unroll
    for (int a = 0; a < 2; ++a)
#pragma unroll
        for (int j = 0; j < 4; ++j)
#pragma unroll
            for (int v = 0; v < 8; ++v) {
                const int row = base + 16 * a + v + 8 * half;
                if (row < N)
                    out[(size_t)row * COUT + 16 * j + m] = acc[a][j][v];
            }
}

// ======================================================================
// Host launcher (graph-capture safe: launches only, all on `stream`)
// ======================================================================
extern "C" void kernel_launch(void* const* d_in, const int* in_sizes, int n_in,
                              void* d_out, int out_size, void* d_ws, size_t ws_size,
                              hipStream_t stream) {
    const float* feat   = (const float*)d_in[0];   // (N, 64) f32
    const int*   coors  = (const int*)  d_in[1];   // (N, 4)  i32
    const float* weight = (const float*)d_in[2];   // (27, 64, 64) f32
    const int*   gmap   = (const int*)  d_in[3];   // (27, 27) i32
    float*       out    = (float*)d_out;

    const int N = in_sizes[0] / CIN;               // 200,000

    // workspace layout: [packed bf16 weights | dense grid]
    char* ws     = (char*)d_ws;
    v16bf* wpack = (v16bf*)ws;                     // 221,184 B
    int*   grid  = (int*)(ws + WPACK_BYTES);       // 52.48 MB

    // 1) grid <- -1
    {
        const int n4 = GRID_CELLS / 4;
        sgc_init_grid<<<(n4 + 255) / 256, 256, 0, stream>>>(grid);
    }
    // 2) scatter voxel ids
    sgc_scatter<<<(N + 255) / 256, 256, 0, stream>>>(coors, grid, N);
    // 3) pack weights into WMMA B-fragment layout (bf16)
    sgc_pack_weights<<<WPACK_TILES, 32, 0, stream>>>(weight, gmap, wpack);
    // 4) main conv: one wave per 32 voxels, 16 waves per block
    {
        const int nWaves = (N + 31) / 32;                  // 6250
        const int wavesPerBlock = CONV_THREADS / 32;       // 16
        const int blocks = (nWaves + wavesPerBlock - 1) / wavesPerBlock;
        sgc_conv_wmma<<<blocks, CONV_THREADS, 0, stream>>>(feat, coors, wpack,
                                                           grid, out, N, nWaves);
    }
}
